// BSM_9904194585245
// MI455X (gfx1250) — matrix operand order repaired and checked by reading, not compiled, not run on hardware
//
#include <hip/hip_runtime.h>
#include <hip/hip_bf16.h>

// ---------------------------------------------------------------------------
// Mamba block forward for gfx1250 (MI455X).
// B=4, L=2048, D=512, E=1024, N=16, K=4, R=32.
// GEMMs via v_wmma_f32_16x16x32_f16 with 32x64 (2x4 tile) register blocking;
// selective scan via 16-lane half-wave recurrences (state n per lane).
// ---------------------------------------------------------------------------

typedef __attribute__((ext_vector_type(8)))  _Float16 v8h;
typedef __attribute__((ext_vector_type(16))) _Float16 v16h;
typedef __attribute__((ext_vector_type(8)))  float    v8f;

#define BQ 4
#define LQ 2048
#define DQ 512
#define EQ 1024
#define NS 16
#define RQ 32
#define MROWS (BQ * LQ)   // 8192

// ---------------------------------------------------------------------------
// Register-blocked WMMA loop: acc[MB][NB] (16x16 f32 tiles) +=
//   A(MB*16 x K, f16 row-major, stride lda) x Bt^T (Bt: NB*16 x K row-major).
// A frag per tile: lane m = lane&15; K halves {k+h*8..+7, k+16+h*8..+7}.
// B frag per tile: lane n = lane&15; K halves {k+h*8..+7, k+16+h*8..+7}.
// ---------------------------------------------------------------------------
template <int MB, int NB>
__device__ __forceinline__ void wmma_block(const _Float16* __restrict__ A, int lda,
                                           const _Float16* __restrict__ Bt, int ldb,
                                           int K, v8f acc[MB][NB]) {
  const int lane = threadIdx.x & 31;
  const int l15  = lane & 15;
  const int half = lane >> 4;
  const _Float16* ar[MB];
  const _Float16* br[NB];
#pragma unroll
  for (int i = 0; i < MB; ++i) ar[i] = A + (size_t)(i * 16 + l15) * lda;
#pragma unroll
  for (int j = 0; j < NB; ++j) br[j] = Bt + (size_t)(j * 16 + l15) * ldb;
#pragma unroll
  for (int i = 0; i < MB; ++i)
#pragma unroll
    for (int j = 0; j < NB; ++j) acc[i][j] = (v8f){};

  for (int k = 0; k < K; k += 32) {
    v16h a[MB], b[NB];
#pragma unroll
    for (int i = 0; i < MB; ++i) {
      ((v8h*)&a[i])[0] = *(const v8h*)(ar[i] + k + half * 8);
      ((v8h*)&a[i])[1] = *(const v8h*)(ar[i] + k + 16 + half * 8);
    }
#pragma unroll
    for (int j = 0; j < NB; ++j) {
      ((v8h*)&b[j])[0] = *(const v8h*)(br[j] + k + half * 8);
      ((v8h*)&b[j])[1] = *(const v8h*)(br[j] + k + 16 + half * 8);
    }
    if (k + 32 < K) {
      __builtin_prefetch(ar[0] + k + 32, 0, 1);   // global_prefetch_b8
      __builtin_prefetch(br[0] + k + 32, 0, 1);
    }
#pragma unroll
    for (int i = 0; i < MB; ++i)
#pragma unroll
      for (int j = 0; j < NB; ++j)
        acc[i][j] = __builtin_amdgcn_wmma_f32_16x16x32_f16(false, a[i], false, b[j],
                                                           (short)0, acc[i][j],
                                                           false, false);
  }
}

// ---------------------------------------------------------------------------
// Kernel 0: convert weights to f16 (layout already [N][K]) + A = -exp(A_log).
// ---------------------------------------------------------------------------
__global__ void prep_kernel(const float* __restrict__ Win, const float* __restrict__ Wxf,
                            const float* __restrict__ Wxr, const float* __restrict__ Wdf,
                            const float* __restrict__ Wdr, const float* __restrict__ Wo,
                            const float* __restrict__ Alf, const float* __restrict__ Alr,
                            _Float16* Win16, _Float16* Wxf16, _Float16* Wxr16,
                            _Float16* Wdf16, _Float16* Wdr16, _Float16* Wo16,
                            float* Anf, float* Anr) {
  size_t i = (size_t)blockIdx.x * blockDim.x + threadIdx.x;
  size_t st = (size_t)gridDim.x * blockDim.x;
  for (size_t j = i; j < (size_t)2 * EQ * DQ; j += st) Win16[j] = (_Float16)Win[j];
  for (size_t j = i; j < (size_t)(RQ + 2 * NS) * EQ; j += st) {
    Wxf16[j] = (_Float16)Wxf[j]; Wxr16[j] = (_Float16)Wxr[j];
  }
  for (size_t j = i; j < (size_t)EQ * RQ; j += st) {
    Wdf16[j] = (_Float16)Wdf[j]; Wdr16[j] = (_Float16)Wdr[j];
  }
  for (size_t j = i; j < (size_t)DQ * EQ; j += st) Wo16[j] = (_Float16)Wo[j];
  for (size_t j = i; j < (size_t)EQ * NS; j += st) {
    Anf[j] = -expf(Alf[j]); Anr[j] = -expf(Alr[j]);
  }
}

// ---------------------------------------------------------------------------
// Kernel 1: LayerNorm of hidden_states -> h16 [(b,l)][512] f16.
// ---------------------------------------------------------------------------
__global__ void layernorm_kernel(const float* __restrict__ hs, const float* __restrict__ nw,
                                 const float* __restrict__ nb, _Float16* __restrict__ h16) {
  __shared__ float red[256];
  const int row = blockIdx.x, tid = threadIdx.x;
  const float* x = hs + (size_t)row * DQ;
  float x0 = x[tid], x1 = x[tid + 256];
  red[tid] = x0 + x1; __syncthreads();
  for (int o = 128; o > 0; o >>= 1) { if (tid < o) red[tid] += red[tid + o]; __syncthreads(); }
  const float mu = red[0] * (1.f / DQ);
  __syncthreads();
  const float d0 = x0 - mu, d1 = x1 - mu;
  red[tid] = d0 * d0 + d1 * d1; __syncthreads();
  for (int o = 128; o > 0; o >>= 1) { if (tid < o) red[tid] += red[tid + o]; __syncthreads(); }
  const float rs = rsqrtf(red[0] * (1.f / DQ) + 1e-5f);
  h16[(size_t)row * DQ + tid]       = (_Float16)(d0 * rs * nw[tid] + nb[tid]);
  h16[(size_t)row * DQ + tid + 256] = (_Float16)(d1 * rs * nw[tid + 256] + nb[tid + 256]);
}

// ---------------------------------------------------------------------------
// Kernel 2: in_proj GEMM: xz[b][n][l] = h16[(b,l)][:] . Win16[n][:]  (K=512)
// Each wave: 32x64 output block (2x4 WMMA tiles).
// ---------------------------------------------------------------------------
__global__ void gemm_xz_kernel(const _Float16* __restrict__ h16,
                               const _Float16* __restrict__ Win16,
                               float* __restrict__ xz) {
  const int NB64 = (2 * EQ) / 64;  // 32 column blocks
  const int wt = blockIdx.x * 8 + (threadIdx.x >> 5);
  if (wt >= (MROWS / 32) * NB64) return;
  const int mt = wt / NB64, nt = wt % NB64;
  v8f acc[2][4];
  wmma_block<2, 4>(h16 + (size_t)mt * 32 * DQ, DQ,
                   Win16 + (size_t)nt * 64 * DQ, DQ, DQ, acc);
  const int lane = threadIdx.x & 31, l15 = lane & 15, half = lane >> 4;
#pragma unroll
  for (int i = 0; i < 2; ++i)
#pragma unroll
    for (int j = 0; j < 4; ++j) {
      const int n = nt * 64 + j * 16 + l15;
#pragma unroll
      for (int r = 0; r < 8; ++r) {
        const int m = mt * 32 + i * 16 + half * 8 + r;
        const int b = m >> 11, l = m & (LQ - 1);
        xz[((size_t)b * (2 * EQ) + n) * LQ + l] = acc[i][j][r];
      }
    }
}

// ---------------------------------------------------------------------------
// Kernel 3: causal depthwise conv (K=4) + SiLU -> xcT[(b,t)][e] f16.
// ---------------------------------------------------------------------------
template <bool REV>
__global__ void conv_silu_kernel(const float* __restrict__ xz, const float* __restrict__ cw,
                                 const float* __restrict__ cb, _Float16* __restrict__ xcT) {
  const int row = blockIdx.y;                 // b*E + e
  const int b = row >> 10, e = row & (EQ - 1);
  const int t = blockIdx.x * blockDim.x + threadIdx.x;
  if (t >= LQ) return;
  const float* x = xz + ((size_t)b * (2 * EQ) + e) * LQ;   // x channels = [0,E)
  const float w0 = cw[e * 4 + 0], w1 = cw[e * 4 + 1], w2 = cw[e * 4 + 2], w3 = cw[e * 4 + 3];
  auto xat = [&](int tt) -> float {
    if (tt < 0) return 0.f;
    return x[REV ? (LQ - 1 - tt) : tt];
  };
  float v = w0 * xat(t - 3) + w1 * xat(t - 2) + w2 * xat(t - 1) + w3 * xat(t) + cb[e];
  v = v / (1.f + expf(-v));                  // SiLU
  xcT[((size_t)b * LQ + t) * EQ + e] = (_Float16)v;
}

// ---------------------------------------------------------------------------
// Kernel 4: x_proj GEMM: xdbl[(b,t)][64] = xcT[(b,t)][:] . Wx[n][:]  (K=1024)
// One wave covers all 64 output columns (2x4 tiles); also emits dt_low f16.
// ---------------------------------------------------------------------------
__global__ void gemm_xdbl_kernel(const _Float16* __restrict__ xcT,
                                 const _Float16* __restrict__ Wx,
                                 float* __restrict__ xdbl, _Float16* __restrict__ dtlow) {
  const int wt = blockIdx.x * 8 + (threadIdx.x >> 5);
  if (wt >= MROWS / 32) return;
  const int mt = wt;
  v8f acc[2][4];
  wmma_block<2, 4>(xcT + (size_t)mt * 32 * EQ, EQ, Wx, EQ, EQ, acc);
  const int lane = threadIdx.x & 31, l15 = lane & 15, half = lane >> 4;
#pragma unroll
  for (int i = 0; i < 2; ++i)
#pragma unroll
    for (int j = 0; j < 4; ++j) {
      const int n = j * 16 + l15;
#pragma unroll
      for (int r = 0; r < 8; ++r) {
        const int m = mt * 32 + i * 16 + half * 8 + r;
        const float v = acc[i][j][r];
        xdbl[(size_t)m * 64 + n] = v;
        if (n < RQ) dtlow[(size_t)m * RQ + n] = (_Float16)v;
      }
    }
}

// ---------------------------------------------------------------------------
// Kernel 5: dt GEMM (K=32, single WMMA step, A reused over 4 N-tiles)
//           + bias + softplus -> delta[(b*E+e)][t] f32.
// ---------------------------------------------------------------------------
__global__ void gemm_dt_kernel(const _Float16* __restrict__ dtlow,
                               const _Float16* __restrict__ Wd,
                               const float* __restrict__ dtb, float* __restrict__ delta) {
  const int NB64 = EQ / 64;  // 16
  const int wt = blockIdx.x * 8 + (threadIdx.x >> 5);
  if (wt >= (MROWS / 16) * NB64) return;
  const int mt = wt / NB64, nt = wt % NB64;
  v8f acc[1][4];
  wmma_block<1, 4>(dtlow + (size_t)mt * 16 * RQ, RQ,
                   Wd + (size_t)nt * 64 * RQ, RQ, RQ, acc);
  const int lane = threadIdx.x & 31, l15 = lane & 15, half = lane >> 4;
#pragma unroll
  for (int j = 0; j < 4; ++j) {
    const int n = nt * 64 + j * 16 + l15;
    const float bias = dtb[n];
#pragma unroll
    for (int r = 0; r < 8; ++r) {
      const int m = mt * 16 + half * 8 + r;
      const int b = m >> 11, t = m & (LQ - 1);
      float v = acc[0][j][r] + bias;
      v = (v > 20.f) ? v : log1pf(expf(v));    // softplus
      delta[((size_t)b * EQ + n) * LQ + t] = v;
    }
  }
}

// ---------------------------------------------------------------------------
// Kernel 6: selective scan. 16 lanes per (b,e) recurrence, state n per lane.
// REV branch reads its reversed-time intermediates and accumulates gated
// output into ysum at original position L-1-t (after fwd kernel wrote it).
// ---------------------------------------------------------------------------
template <bool REV>
__global__ void scan_kernel(const float* __restrict__ delta, const _Float16* __restrict__ xcT,
                            const float* __restrict__ xdbl, const float* __restrict__ Aneg,
                            const float* __restrict__ Dv, const float* __restrict__ xz,
                            float* __restrict__ ysum) {
  const int grp = threadIdx.x >> 4, lane16 = threadIdx.x & 15;
  const int pair = blockIdx.x * 16 + grp;          // b*E + e  (grid sized exactly)
  const int b = pair >> 10, e = pair & (EQ - 1);
  const float a   = Aneg[e * NS + lane16];
  const float dve = Dv[e];
  const float* dptr  = delta + (size_t)pair * LQ;
  const float* zbase = xz + ((size_t)b * (2 * EQ) + EQ + e) * LQ;
  float h = 0.f;
  for (int t = 0; t < LQ; ++t) {
    const float dt = dptr[t];
    const float u  = (float)xcT[((size_t)b * LQ + t) * EQ + e];
    const size_t xb = ((size_t)b * LQ + t) * 64;
    const float bm = xdbl[xb + RQ + lane16];
    const float cm = xdbl[xb + RQ + NS + lane16];
    const float dA = __expf(dt * a);
    h = fmaf(dA, h, (dt * u) * bm);
    float y = h * cm;
    y += __shfl_xor(y, 1, 16);
    y += __shfl_xor(y, 2, 16);
    y += __shfl_xor(y, 4, 16);
    y += __shfl_xor(y, 8, 16);
    if (lane16 == 0) {
      const int lo = REV ? (LQ - 1 - t) : t;
      const float z = zbase[lo];
      const float yy = y + dve * u;
      const float out = yy * (z / (1.f + expf(-z)));   // y * silu(z)
      const size_t oi = ((size_t)b * LQ + lo) * EQ + e;
      if (REV) ysum[oi] += out; else ysum[oi] = out;
    }
  }
}

// ---------------------------------------------------------------------------
// Kernel 7: ysum f32 -> f16 for out_proj GEMM.
// ---------------------------------------------------------------------------
__global__ void cvt_y16_kernel(const float* __restrict__ ysum, _Float16* __restrict__ y16) {
  size_t i = (size_t)blockIdx.x * blockDim.x + threadIdx.x;
  size_t st = (size_t)gridDim.x * blockDim.x;
  for (size_t j = i; j < (size_t)MROWS * EQ; j += st) y16[j] = (_Float16)ysum[j];
}

// ---------------------------------------------------------------------------
// Kernel 8: out_proj GEMM: yproj[(b,l)][d] = y16[(b,l)][:] . Wo[d][:]  (K=1024)
// ---------------------------------------------------------------------------
__global__ void gemm_out_kernel(const _Float16* __restrict__ y16,
                                const _Float16* __restrict__ Wo,
                                float* __restrict__ yproj) {
  const int NB64 = DQ / 64;  // 8
  const int wt = blockIdx.x * 8 + (threadIdx.x >> 5);
  if (wt >= (MROWS / 32) * NB64) return;
  const int mt = wt / NB64, nt = wt % NB64;
  v8f acc[2][4];
  wmma_block<2, 4>(y16 + (size_t)mt * 32 * EQ, EQ,
                   Wo + (size_t)nt * 64 * EQ, EQ, EQ, acc);
  const int lane = threadIdx.x & 31, l15 = lane & 15, half = lane >> 4;
#pragma unroll
  for (int i = 0; i < 2; ++i)
#pragma unroll
    for (int j = 0; j < 4; ++j) {
      const int n = nt * 64 + j * 16 + l15;
#pragma unroll
      for (int r = 0; r < 8; ++r) {
        const int m = mt * 32 + i * 16 + half * 8 + r;
        yproj[(size_t)m * DQ + n] = acc[i][j][r];
      }
    }
}

// ---------------------------------------------------------------------------
// Kernel 9: residual add + RMSNorm -> d_out.
// ---------------------------------------------------------------------------
__global__ void rmsnorm_kernel(const float* __restrict__ yproj, const float* __restrict__ hs,
                               const float* __restrict__ w, float* __restrict__ out) {
  __shared__ float red[256];
  const int row = blockIdx.x, tid = threadIdx.x;
  const size_t base = (size_t)row * DQ;
  const float v0 = yproj[base + tid]       + hs[base + tid];
  const float v1 = yproj[base + tid + 256] + hs[base + tid + 256];
  red[tid] = v0 * v0 + v1 * v1; __syncthreads();
  for (int o = 128; o > 0; o >>= 1) { if (tid < o) red[tid] += red[tid + o]; __syncthreads(); }
  const float rs = rsqrtf(red[0] * (1.f / DQ) + 1e-5f);
  out[base + tid]       = v0 * rs * w[tid];
  out[base + tid + 256] = v1 * rs * w[tid + 256];
}

// ---------------------------------------------------------------------------
extern "C" void kernel_launch(void* const* d_in, const int* in_sizes, int n_in,
                              void* d_out, int out_size, void* d_ws, size_t ws_size,
                              hipStream_t stream) {
  (void)in_sizes; (void)n_in; (void)out_size; (void)ws_size;
  const float* hs    = (const float*)d_in[0];
  const float* nw    = (const float*)d_in[1];
  const float* nb    = (const float*)d_in[2];
  const float* Win   = (const float*)d_in[3];
  const float* cwf   = (const float*)d_in[4];
  const float* cbf   = (const float*)d_in[5];
  const float* Wxf   = (const float*)d_in[6];
  const float* Wdf   = (const float*)d_in[7];
  const float* dtbf  = (const float*)d_in[8];
  const float* Alf   = (const float*)d_in[9];
  const float* Dfw   = (const float*)d_in[10];
  const float* cwr   = (const float*)d_in[11];
  const float* cbr   = (const float*)d_in[12];
  const float* Wxr   = (const float*)d_in[13];
  const float* Wdr   = (const float*)d_in[14];
  const float* dtbr  = (const float*)d_in[15];
  const float* Alr   = (const float*)d_in[16];
  const float* Dbw   = (const float*)d_in[17];
  const float* Wo    = (const float*)d_in[18];
  const float* wf    = (const float*)d_in[19];
  float* out = (float*)d_out;

  char* p = (char*)d_ws;
  auto alloc = [&](size_t bytes) -> void* {
    void* r = (void*)p;
    p += (bytes + 255) & ~(size_t)255;
    return r;
  };
  _Float16* h16   = (_Float16*)alloc((size_t)MROWS * DQ * 2);
  _Float16* Win16 = (_Float16*)alloc((size_t)2 * EQ * DQ * 2);
  _Float16* Wxf16 = (_Float16*)alloc((size_t)64 * EQ * 2);
  _Float16* Wxr16 = (_Float16*)alloc((size_t)64 * EQ * 2);
  _Float16* Wdf16 = (_Float16*)alloc((size_t)EQ * RQ * 2);
  _Float16* Wdr16 = (_Float16*)alloc((size_t)EQ * RQ * 2);
  _Float16* Wo16  = (_Float16*)alloc((size_t)DQ * EQ * 2);
  float*    Anf   = (float*)alloc((size_t)EQ * NS * 4);
  float*    Anr   = (float*)alloc((size_t)EQ * NS * 4);
  float*    xz    = (float*)alloc((size_t)BQ * 2 * EQ * LQ * 4);
  _Float16* xcTf  = (_Float16*)alloc((size_t)MROWS * EQ * 2);
  _Float16* xcTr  = (_Float16*)alloc((size_t)MROWS * EQ * 2);
  float*    xdblf = (float*)alloc((size_t)MROWS * 64 * 4);
  float*    xdblr = (float*)alloc((size_t)MROWS * 64 * 4);
  _Float16* dtlf  = (_Float16*)alloc((size_t)MROWS * RQ * 2);
  _Float16* dtlr  = (_Float16*)alloc((size_t)MROWS * RQ * 2);
  float*    delf  = (float*)alloc((size_t)BQ * EQ * LQ * 4);
  float*    delr  = (float*)alloc((size_t)BQ * EQ * LQ * 4);
  float*    ysum  = (float*)alloc((size_t)MROWS * EQ * 4);
  _Float16* y16   = (_Float16*)alloc((size_t)MROWS * EQ * 2);
  float*    yproj = (float*)alloc((size_t)MROWS * DQ * 4);

  // 0) weight conversion
  prep_kernel<<<1024, 256, 0, stream>>>(Win, Wxf, Wxr, Wdf, Wdr, Wo, Alf, Alr,
                                        Win16, Wxf16, Wxr16, Wdf16, Wdr16, Wo16, Anf, Anr);
  // 1) layernorm
  layernorm_kernel<<<MROWS, 256, 0, stream>>>(hs, nw, nb, h16);
  // 2) in_proj GEMM: (8192/32)*(2048/64) = 8192 wave-tiles, 8 waves/block
  gemm_xz_kernel<<<(MROWS / 32) * ((2 * EQ) / 64) / 8, 256, 0, stream>>>(h16, Win16, xz);
  // 3) conv + silu, both branches
  conv_silu_kernel<false><<<dim3(LQ / 256, BQ * EQ), 256, 0, stream>>>(xz, cwf, cbf, xcTf);
  conv_silu_kernel<true ><<<dim3(LQ / 256, BQ * EQ), 256, 0, stream>>>(xz, cwr, cbr, xcTr);
  // 4) x_proj GEMMs: 256 wave-tiles
  gemm_xdbl_kernel<<<(MROWS / 32) / 8, 256, 0, stream>>>(xcTf, Wxf16, xdblf, dtlf);
  gemm_xdbl_kernel<<<(MROWS / 32) / 8, 256, 0, stream>>>(xcTr, Wxr16, xdblr, dtlr);
  // 5) dt GEMM + softplus: (8192/16)*(1024/64) = 8192 wave-tiles
  gemm_dt_kernel<<<(MROWS / 16) * (EQ / 64) / 8, 256, 0, stream>>>(dtlf, Wdf16, dtbf, delf);
  gemm_dt_kernel<<<(MROWS / 16) * (EQ / 64) / 8, 256, 0, stream>>>(dtlr, Wdr16, dtbr, delr);
  // 6) selective scans (fwd writes, rev accumulates)
  scan_kernel<false><<<(BQ * EQ) / 16, 256, 0, stream>>>(delf, xcTf, xdblf, Anf, Dfw, xz, ysum);
  scan_kernel<true ><<<(BQ * EQ) / 16, 256, 0, stream>>>(delr, xcTr, xdblr, Anr, Dbw, xz, ysum);
  // 7) convert to f16
  cvt_y16_kernel<<<2048, 256, 0, stream>>>(ysum, y16);
  // 8) out_proj GEMM: (8192/32)*(512/64) = 2048 wave-tiles
  gemm_out_kernel<<<(MROWS / 32) * (DQ / 64) / 8, 256, 0, stream>>>(y16, Wo16, yproj);
  // 9) residual + RMSNorm
  rmsnorm_kernel<<<MROWS, 256, 0, stream>>>(yproj, hs, wf, out);
}
